// attnVec_topK_61546881351806
// MI455X (gfx1250) — matrix operand order repaired and checked by reading, not compiled, not run on hardware
//
#include <hip/hip_runtime.h>
#include <hip/hip_bf16.h>
#include <cmath>

#define NP_PATHS 8
#define DDIM 128
#define TOPK 4
#define QPB 32           // questions per block
#define ROWS_PB 256      // rows (n,p pairs) per block

typedef __attribute__((ext_vector_type(16))) __bf16 v16bf;
typedef __attribute__((ext_vector_type(8)))  float  v8f;

__device__ __forceinline__ float fast_tanh(float x) {
#if __has_builtin(__builtin_amdgcn_tanhf)
  return __builtin_amdgcn_tanhf(x);
#elif __has_builtin(__builtin_amdgcn_tanh_f32)
  return __builtin_amdgcn_tanh_f32(x);
#else
  return tanhf(x);
#endif
}

__device__ __forceinline__ v16bf cvt16(float4 a, float4 b, float4 c, float4 d) {
  v16bf r;
  r[0]  = (__bf16)a.x; r[1]  = (__bf16)a.y; r[2]  = (__bf16)a.z; r[3]  = (__bf16)a.w;
  r[4]  = (__bf16)b.x; r[5]  = (__bf16)b.y; r[6]  = (__bf16)b.z; r[7]  = (__bf16)b.w;
  r[8]  = (__bf16)c.x; r[9]  = (__bf16)c.y; r[10] = (__bf16)c.z; r[11] = (__bf16)c.w;
  r[12] = (__bf16)d.x; r[13] = (__bf16)d.y; r[14] = (__bf16)d.z; r[15] = (__bf16)d.w;
  return r;
}

__global__ __launch_bounds__(256) void han_attn_topk_kernel(
    const float* __restrict__ emb,   // [N, 8, 128]
    const float* __restrict__ fcw,   // [128, 128]
    const float* __restrict__ fcb,   // [128]
    const float* __restrict__ attn,  // [128]
    float* __restrict__ out,         // [N, 128]
    int nQ) {
  // W staged in LDS as bf16 in exact B-matrix per-lane chunks:
  // chunk index c = e0*128 + kk*32 + lane, 16 bf16 (32B) per chunk.
  __shared__ __bf16 sW[8 * 4 * 32 * 16];   // 32 KB
  __shared__ float  sBias[DDIM];
  __shared__ float  sAttn[DDIM];
  __shared__ float  sScores[QPB * NP_PATHS];

  const int tid = threadIdx.x;

  // ---------------- Phase 1: stage W / bias / attn into LDS ----------------
  #pragma unroll
  for (int i = 0; i < 4; ++i) {
    const int c     = tid + i * 256;        // [0, 1024)
    const int laneC = c & 31;
    const int kkC   = (c >> 5) & 3;
    const int e0C   = c >> 7;
    const int wrow  = e0C * 16 + (laneC & 15);           // N column of B
    const int kb    = kkC * 32 + (laneC >> 4) * 16;      // K base for this lane half
    const float4* src = (const float4*)(fcw + wrow * DDIM + kb);
    v16bf v = cvt16(src[0], src[1], src[2], src[3]);
    *(v16bf*)(&sW[c * 16]) = v;
  }
  if (tid < DDIM) { sBias[tid] = fcb[tid]; sAttn[tid] = attn[tid]; }
  __syncthreads();

  // ---------------- Phase 2: bf16 WMMA projection + score --------------------
  const int lane    = tid & 31;
  const int wv      = tid >> 5;
  const int halfSel = lane >> 4;
  const int lm      = lane & 15;

  const long long NProws  = (long long)nQ * NP_PATHS;
  const long long rowBase = (long long)blockIdx.x * ROWS_PB + (long long)wv * 32;

  // A tiles: 2 tiles x 16 rows, 16x32 per K-step, contiguous k = 16*halfSel + s.
  v16bf A0[4], A1[4];
  {
    long long r0 = rowBase + lm;
    long long r1 = rowBase + 16 + lm;
    if (r0 >= NProws) r0 = NProws - 1;   // safety clamp (exact fit for N=100000)
    if (r1 >= NProws) r1 = NProws - 1;
    #pragma unroll
    for (int kk = 0; kk < 4; ++kk) {
      const int kb = kk * 32 + halfSel * 16;
      const float4* p0 = (const float4*)(emb + r0 * DDIM + kb);
      const float4* p1 = (const float4*)(emb + r1 * DDIM + kb);
      A0[kk] = cvt16(p0[0], p0[1], p0[2], p0[3]);   // K: kb..kb+15
      A1[kk] = cvt16(p1[0], p1[1], p1[2], p1[3]);
    }
  }

  v8f sp0 = {}; v8f sp1 = {};
  #pragma unroll
  for (int e0 = 0; e0 < 8; ++e0) {
    v8f acc0 = {}; v8f acc1 = {};
    #pragma unroll
    for (int kk = 0; kk < 4; ++kk) {
      v16bf b = *(const v16bf*)(&sW[(e0 * 128 + kk * 32 + lane) * 16]);
      acc0 = __builtin_amdgcn_wmma_f32_16x16x32_bf16(false, A0[kk], false, b,
                                                     (short)0, acc0, false, false);
      acc1 = __builtin_amdgcn_wmma_f32_16x16x32_bf16(false, A1[kk], false, b,
                                                     (short)0, acc1, false, false);
    }
    // C layout: vgpr j -> row M=j+8*halfSel, lane%16 -> column
    const int   col  = e0 * 16 + lm;
    const float bias = sBias[col];
    const float av   = sAttn[col];
    #pragma unroll
    for (int j = 0; j < 8; ++j) {
      sp0[j] += fast_tanh(acc0[j] + bias) * av;
      sp1[j] += fast_tanh(acc1[j] + bias) * av;
    }
  }

  // Reduce the 16 column-lanes within each half-wave.
  #pragma unroll
  for (int m = 8; m >= 1; m >>= 1) {
    #pragma unroll
    for (int j = 0; j < 8; ++j) {
      sp0[j] += __shfl_xor(sp0[j], m, 32);
      sp1[j] += __shfl_xor(sp1[j], m, 32);
    }
  }

  if (lm == 0) {   // lanes 0 and 16 hold the two 8-row groups of each tile
    const int q0 = wv * 4 + halfSel;       // tile 0 question slot in block
    const int q1 = wv * 4 + 2 + halfSel;   // tile 1
    #pragma unroll
    for (int j = 0; j < 8; ++j) {
      sScores[q0 * 8 + j] = sp0[j];
      sScores[q1 * 8 + j] = sp1[j];
    }
  }
  __syncthreads();

  // ---------------- Phase 3: softmax, top-4, weighted gather-sum ------------
  const int ql  = tid >> 3;   // question slot [0,32)
  const int seg = tid & 7;    // 8 threads/question, 16 floats each
  const long long qg = (long long)blockIdx.x * QPB + ql;
  if (qg < nQ) {
    float s[8];
    #pragma unroll
    for (int p = 0; p < 8; ++p) s[p] = sScores[ql * 8 + p];
    float mx = s[0];
    #pragma unroll
    for (int p = 1; p < 8; ++p) mx = fmaxf(mx, s[p]);
    float w8[8]; float sum = 0.f;
    #pragma unroll
    for (int p = 0; p < 8; ++p) { w8[p] = __expf(s[p] - mx); sum += w8[p]; }
    const float inv = 1.0f / sum;
    #pragma unroll
    for (int p = 0; p < 8; ++p) w8[p] *= inv;

    float4 oacc[4];
    #pragma unroll
    for (int j = 0; j < 4; ++j) oacc[j] = make_float4(0.f, 0.f, 0.f, 0.f);

    #pragma unroll
    for (int k = 0; k < TOPK; ++k) {
      int best = 0; float bv = -1.0f;   // softmax weights are >= 0
      #pragma unroll
      for (int p = 0; p < 8; ++p) {
        if (w8[p] > bv) { bv = w8[p]; best = p; }   // strict '>' == jax tie-break
      }
      w8[best] = -2.0f;
      const float4* src = (const float4*)(emb + (qg * 8 + best) * DDIM + seg * 16);
      #pragma unroll
      for (int j = 0; j < 4; ++j) {
        float4 v = src[j];
        oacc[j].x += bv * v.x; oacc[j].y += bv * v.y;
        oacc[j].z += bv * v.z; oacc[j].w += bv * v.w;
      }
    }
    float4* dst = (float4*)(out + qg * DDIM + seg * 16);
    #pragma unroll
    for (int j = 0; j < 4; ++j) dst[j] = oacc[j];
  }
}

extern "C" void kernel_launch(void* const* d_in, const int* in_sizes, int n_in,
                              void* d_out, int out_size, void* d_ws, size_t ws_size,
                              hipStream_t stream) {
  const float* emb  = (const float*)d_in[0];
  const float* fcw  = (const float*)d_in[1];
  const float* fcb  = (const float*)d_in[2];
  const float* attn = (const float*)d_in[3];
  float* out = (float*)d_out;

  const int nQ = in_sizes[0] / (NP_PATHS * DDIM);   // 100000
  const int blocks = (nQ + QPB - 1) / QPB;          // 3125, exact fit
  hipLaunchKernelGGL(han_attn_topk_kernel, dim3(blocks), dim3(256), 0, stream,
                     emb, fcw, fcb, attn, out, nQ);
}
